// AttentionNCF_37039797960808
// MI455X (gfx1250) — compile-verified, matrix operand order checked by reading, not run on hardware
//
#include <hip/hip_runtime.h>

// ---------------------------------------------------------------------------
// AttentionNCF forward for MI455X (gfx1250, wave32, WMMA bf16, bf16 data path)
// B=1024, I=4096, D=512, IE=256, UE=512, D1..D4 = 1024/512/256/128
// ---------------------------------------------------------------------------

#define BQ   1024   // batch rows
#define II   4096   // rated items
#define DD   512    // embedding dim

#define BM   128
#define BN   128
#define BK   32
#define APAD 40     // bf16 elems per LDS row: 32 data + 8 pad (80B, 16B-aligned rows)

typedef __attribute__((ext_vector_type(16))) __bf16       v16bf;
typedef __attribute__((ext_vector_type(8)))  float        v8f;
typedef __attribute__((ext_vector_type(4)))  unsigned int u32x4;
typedef __attribute__((ext_vector_type(4)))  float        f32x4;
typedef unsigned short ushort_t;

union FragAB { v16bf v; u32x4 q[2]; };
union FragC  { v8f   v; float f[8]; };
union Pack8  { u32x4 q; ushort_t h[8]; };

__device__ __forceinline__ ushort_t f32_to_bf16(float f) {
  unsigned int u = __float_as_uint(f);
  u += 0x7FFFu + ((u >> 16) & 1u);      // round-to-nearest-even
  return (ushort_t)(u >> 16);
}
__device__ __forceinline__ float bf16_to_f32(ushort_t h) {
  return __uint_as_float(((unsigned int)h) << 16);
}

// ---------------------------------------------------------------------------
// One-shot packed f32 -> bf16 conversion (8 elements / thread, b128 store)
// ---------------------------------------------------------------------------
__global__ __launch_bounds__(256)
void cvt_f32_bf16(const float* __restrict__ src, ushort_t* __restrict__ dst, int n8)
{
  int i = blockIdx.x * 256 + threadIdx.x;
  if (i >= n8) return;
  const f32x4 a = *(const f32x4*)(src + (size_t)i * 8);
  const f32x4 b = *(const f32x4*)(src + (size_t)i * 8 + 4);
  u32x4 o;
  o[0] = (unsigned)f32_to_bf16(a[0]) | ((unsigned)f32_to_bf16(a[1]) << 16);
  o[1] = (unsigned)f32_to_bf16(a[2]) | ((unsigned)f32_to_bf16(a[3]) << 16);
  o[2] = (unsigned)f32_to_bf16(b[0]) | ((unsigned)f32_to_bf16(b[1]) << 16);
  o[3] = (unsigned)f32_to_bf16(b[2]) | ((unsigned)f32_to_bf16(b[3]) << 16);
  *(u32x4*)(dst + (size_t)i * 8) = o;
}

// ---------------------------------------------------------------------------
// Tiled bf16 WMMA GEMM: C[M x N](bf16) = act( A[M x K](bf16) @ Bw[K x N](bf16)
//                                             + bias(f32) ),  f32 accumulate.
// Block = 256 threads = 8 waves; block tile 128x128; wave tile 32x64
// (2 M-frags x 4 N-frags = 8 WMMA per wave per K-tile).
// Software-pipelined: next tile's global b128 loads issued before the WMMAs.
// ---------------------------------------------------------------------------
__global__ __launch_bounds__(256)
void gemm_bf16_wmma(const ushort_t* __restrict__ A, const ushort_t* __restrict__ Bw,
                    ushort_t* __restrict__ C, const float* __restrict__ bias,
                    int N, int K, int ldc, int relu)
{
  __shared__ __align__(16) ushort_t As[BM * APAD];  // M x K tile
  __shared__ __align__(16) ushort_t Bs[BN * APAD];  // N x K tile (transposed)

  const int tid  = threadIdx.x;
  const int lane = tid & 31;
  const int wave = tid >> 5;
  const int wm   = (wave >> 1) * 32;      // wave row inside block tile (0..96)
  const int wn   = (wave & 1)  * 64;      // wave col inside block tile (0/64)
  const int bm   = blockIdx.y * BM;
  const int bn   = blockIdx.x * BN;

  const int lm = lane & 15;               // row/col within 16-wide frag
  const int lh = lane >> 4;               // lane half selects K sub-chunk

  // ---- staging maps ----
  // A: 128x32 bf16 = 512 b128 chunks, 2 per thread (pure copies).
  const int r0 = tid >> 2,         c0 = (tid & 3) << 3;
  const int r1 = (tid + 256) >> 2, c1 = (tid & 3) << 3;
  // B: 32x128 bf16; each thread loads 2 adjacent K-rows at one N-offset so the
  // transposed values pack into (k,k+1) dwords -> 8 ds_store_b32.
  const int kp = (tid >> 4) << 1;         // 0,2,..,30
  const int n8 = (tid & 15) << 3;         // 0..120

  const ushort_t* pa0 = A  + (size_t)(bm + r0) * K + c0;
  const ushort_t* pa1 = A  + (size_t)(bm + r1) * K + c1;
  const ushort_t* pb0 = Bw + (size_t)kp * N + bn + n8;
  const ushort_t* pb1 = pb0 + N;

  FragC acc[2][4];
#pragma unroll
  for (int i = 0; i < 2; ++i)
#pragma unroll
    for (int j = 0; j < 4; ++j)
#pragma unroll
      for (int r = 0; r < 8; ++r) acc[i][j].f[r] = 0.0f;

  // preload tile 0 into registers
  u32x4 sa0 = *(const u32x4*)pa0;
  u32x4 sa1 = *(const u32x4*)pa1;
  u32x4 sb0 = *(const u32x4*)pb0;
  u32x4 sb1 = *(const u32x4*)pb1;

  const int kTiles = K / BK;
  for (int kt = 0; kt < kTiles; ++kt) {
    __syncthreads();                       // previous tile's LDS reads complete

    // ---- registers -> LDS ----
    *(u32x4*)&As[r0 * APAD + c0] = sa0;
    *(u32x4*)&As[r1 * APAD + c1] = sa1;
    Pack8 tb0, tb1; tb0.q = sb0; tb1.q = sb1;
#pragma unroll
    for (int j = 0; j < 8; ++j) {          // transpose, packed pair stores
      unsigned v = (unsigned)tb0.h[j] | ((unsigned)tb1.h[j] << 16);
      *(unsigned*)&Bs[(n8 + j) * APAD + kp] = v;
    }

    __syncthreads();                       // tile visible to all waves

    // ---- issue next tile's global loads (overlap with WMMA below) ----
    if (kt + 1 < kTiles) {
      pa0 += BK; pa1 += BK;
      pb0 += (size_t)BK * N; pb1 += (size_t)BK * N;
      sa0 = *(const u32x4*)pa0;
      sa1 = *(const u32x4*)pa1;
      sb0 = *(const u32x4*)pb0;
      sb1 = *(const u32x4*)pb1;
    }

    // ---- fragments: ISA 16-bit A/B layout = two contiguous 16B chunks/lane ----
    // lanes 0-15 : K 0-7  -> v[0:3], K 16-23 -> v[4:7]
    // lanes 16-31: K 8-15 -> v[0:3], K 24-31 -> v[4:7]
    FragAB af[2], bf[4];
#pragma unroll
    for (int s = 0; s < 2; ++s) {
      const char* pa = (const char*)&As[(wm + s * 16 + lm) * APAD] + lh * 16;
      af[s].q[0] = *(const u32x4*)(pa);
      af[s].q[1] = *(const u32x4*)(pa + 32);
    }
#pragma unroll
    for (int s = 0; s < 4; ++s) {
      const char* pbs = (const char*)&Bs[(wn + s * 16 + lm) * APAD] + lh * 16;
      bf[s].q[0] = *(const u32x4*)(pbs);
      bf[s].q[1] = *(const u32x4*)(pbs + 32);
    }

#pragma unroll
    for (int i = 0; i < 2; ++i)
#pragma unroll
      for (int j = 0; j < 4; ++j)
        acc[i][j].v = __builtin_amdgcn_wmma_f32_16x16x32_bf16(
            false, af[i].v, false, bf[j].v, (short)0, acc[i][j].v, false, false);
  }

  // ---- epilogue: bias + optional ReLU, bf16 store ----
#pragma unroll
  for (int i = 0; i < 2; ++i)
#pragma unroll
    for (int j = 0; j < 4; ++j) {
      int col  = bn + wn + j * 16 + lm;
      float bv = bias ? bias[col] : 0.0f;
#pragma unroll
      for (int r = 0; r < 8; ++r) {
        int row = bm + wm + i * 16 + r + lh * 8;   // C/D layout: VGPR r -> M=r (+8 hi half)
        float v = acc[i][j].f[r] + bv;
        if (relu) v = fmaxf(v, 0.0f);
        C[(size_t)row * ldc + col] = f32_to_bf16(v);
      }
    }
}

// ---------------------------------------------------------------------------
// Per-row dot product, f32 input (one wave32 per row)
// ---------------------------------------------------------------------------
__global__ __launch_bounds__(32)
void rowdot_kernel(const float* __restrict__ X, const float* __restrict__ w,
                   float* __restrict__ out, int K)
{
  const int row  = blockIdx.x;
  const int lane = threadIdx.x;
  const float* xr = X + (size_t)row * K;
  float s = 0.0f;
  for (int k = lane; k < K; k += 32) s += xr[k] * w[k];
#pragma unroll
  for (int off = 16; off > 0; off >>= 1) s += __shfl_down(s, off, 32);
  if (lane == 0) out[row] = s;
}

// ---------------------------------------------------------------------------
// Final D4 -> 1 layer: bf16 activations . f32 weights + bias
// ---------------------------------------------------------------------------
__global__ __launch_bounds__(32)
void rowdot_bf16_kernel(const ushort_t* __restrict__ X, const float* __restrict__ w,
                        const float* __restrict__ bias, float* __restrict__ out, int K)
{
  const int row  = blockIdx.x;
  const int lane = threadIdx.x;
  const ushort_t* xr = X + (size_t)row * K;
  float s = 0.0f;
  for (int k = lane; k < K; k += 32) s += bf16_to_f32(xr[k]) * w[k];
#pragma unroll
  for (int off = 16; off > 0; off >>= 1) s += __shfl_down(s, off, 32);
  if (lane == 0) out[row] = s + bias[0];
}

// ---------------------------------------------------------------------------
// Masked softmax row, folds in user_matrix, writes bf16:
//   W[b,i] = softmax_i( cs[b]+rs[i]+b_att over um!=0 ) * um[b,i]   (0 if masked)
// ---------------------------------------------------------------------------
__global__ __launch_bounds__(256)
void attn_weights_kernel(const float* __restrict__ cs, const float* __restrict__ rs,
                         const float* __restrict__ um, const float* __restrict__ batt,
                         ushort_t* __restrict__ W)
{
  __shared__ float red[8];
  const int b    = blockIdx.x;
  const int tid  = threadIdx.x;
  const int lane = tid & 31;
  const int wv   = tid >> 5;
  const float cb = cs[b] + batt[0];
  const float* urow = um + (size_t)b * II;
  ushort_t* wrow = W + (size_t)b * II;

  float ls[16], lu[16];
  float m = -INFINITY;
#pragma unroll
  for (int t = 0; t < II / 256; ++t) {
    int i = tid + t * 256;
    float u = urow[i];
    float s = cb + rs[i];
    lu[t] = u; ls[t] = s;
    if (u != 0.0f) m = fmaxf(m, s);
  }
#pragma unroll
  for (int off = 16; off > 0; off >>= 1) m = fmaxf(m, __shfl_xor(m, off, 32));
  if (lane == 0) red[wv] = m;
  __syncthreads();
  float gm = -INFINITY;
#pragma unroll
  for (int i = 0; i < 8; ++i) gm = fmaxf(gm, red[i]);

  float ssum = 0.0f;
#pragma unroll
  for (int t = 0; t < II / 256; ++t) {
    float e = (lu[t] != 0.0f) ? __expf(ls[t] - gm) : 0.0f;
    ls[t] = e;
    ssum += e;
  }
#pragma unroll
  for (int off = 16; off > 0; off >>= 1) ssum += __shfl_xor(ssum, off, 32);
  __syncthreads();
  if (lane == 0) red[wv] = ssum;
  __syncthreads();
  float tot = 0.0f;
#pragma unroll
  for (int i = 0; i < 8; ++i) tot += red[i];
  const float inv = (tot > 0.0f) ? 1.0f / tot : 0.0f;   // fully-masked row -> zeros

#pragma unroll
  for (int t = 0; t < II / 256; ++t) {
    int i = tid + t * 256;
    wrow[i] = f32_to_bf16(ls[t] * inv * lu[t]);
  }
}

// ---------------------------------------------------------------------------
extern "C" void kernel_launch(void* const* d_in, const int* in_sizes, int n_in,
                              void* d_out, int out_size, void* d_ws, size_t ws_size,
                              hipStream_t stream)
{
  const float* cand  = (const float*)d_in[0];   // (1024,512)
  const float* rated = (const float*)d_in[1];   // (4096,512)
  const float* um    = (const float*)d_in[2];   // (1024,4096)
  const float* w_att = (const float*)d_in[3];   // (1024,1)
  const float* b_att = (const float*)d_in[4];   // (1,)
  const float* ie_w1 = (const float*)d_in[5];   const float* ie_b1 = (const float*)d_in[6];
  const float* ie_w2 = (const float*)d_in[7];   const float* ie_b2 = (const float*)d_in[8];
  const float* ue_w1 = (const float*)d_in[9];   const float* ue_b1 = (const float*)d_in[10];
  const float* ue_w2 = (const float*)d_in[11];  const float* ue_b2 = (const float*)d_in[12];
  const float* m_w1  = (const float*)d_in[13];  const float* m_b1  = (const float*)d_in[14];
  const float* m_w2  = (const float*)d_in[15];  const float* m_b2  = (const float*)d_in[16];
  const float* m_w3  = (const float*)d_in[17];  const float* m_b3  = (const float*)d_in[18];
  const float* m_w4  = (const float*)d_in[19];  const float* m_b4  = (const float*)d_in[20];
  const float* m_w5  = (const float*)d_in[21];  const float* m_b5  = (const float*)d_in[22];
  float* out = (float*)d_out;                   // (1024,1)
  (void)in_sizes; (void)n_in; (void)out_size; (void)ws_size;

  // ---- workspace carve-up (256B-aligned slabs) ----
  char* base = (char*)d_ws;
  auto alloc = [&](size_t bytes) -> char* {
    char* p = base; base += (bytes + 255) & ~(size_t)255; return p;
  };
  float*    cs   = (float*)   alloc(BQ * 4);
  float*    rs   = (float*)   alloc(II * 4);
  ushort_t* Wb   = (ushort_t*)alloc((size_t)BQ * II * 2);      // attn*user  (bf16, 8MB)
  ushort_t* rtb  = (ushort_t*)alloc((size_t)II * DD * 2);      // rated  bf16
  ushort_t* cdb  = (ushort_t*)alloc((size_t)BQ * DD * 2);      // cand   bf16
  ushort_t* w1i  = (ushort_t*)alloc((size_t)512 * 512 * 2);
  ushort_t* w2i  = (ushort_t*)alloc((size_t)512 * 256 * 2);
  ushort_t* w1u  = (ushort_t*)alloc((size_t)512 * 1024 * 2);
  ushort_t* w2u  = (ushort_t*)alloc((size_t)1024 * 512 * 2);
  ushort_t* mw1  = (ushort_t*)alloc((size_t)768 * 1024 * 2);
  ushort_t* mw2  = (ushort_t*)alloc((size_t)1024 * 512 * 2);
  ushort_t* mw3  = (ushort_t*)alloc((size_t)512 * 256 * 2);
  ushort_t* mw4  = (ushort_t*)alloc((size_t)256 * 128 * 2);
  ushort_t* ue   = (ushort_t*)alloc((size_t)BQ * 512 * 2);     // user_est
  ushort_t* it1  = (ushort_t*)alloc((size_t)BQ * 512 * 2);
  ushort_t* ut1  = (ushort_t*)alloc((size_t)BQ * 1024 * 2);
  ushort_t* hcat = (ushort_t*)alloc((size_t)BQ * 768 * 2);     // [item_emb | user_emb]
  ushort_t* h1   = (ushort_t*)alloc((size_t)BQ * 1024 * 2);
  ushort_t* h2   = (ushort_t*)alloc((size_t)BQ * 512 * 2);
  ushort_t* h3   = (ushort_t*)alloc((size_t)BQ * 256 * 2);
  ushort_t* h4   = (ushort_t*)alloc((size_t)BQ * 128 * 2);

  auto cvt = [&](const float* s, ushort_t* d, size_t n) {
    int n8 = (int)(n / 8);
    cvt_f32_bf16<<<(n8 + 255) / 256, 256, 0, stream>>>(s, d, n8);
  };

  // 1) attention pre-scores (f32 inputs)
  rowdot_kernel<<<BQ, 32, 0, stream>>>(cand,  w_att,      cs, DD);
  rowdot_kernel<<<II, 32, 0, stream>>>(rated, w_att + DD, rs, DD);

  // 2) one-shot bf16 conversions of all GEMM operands
  cvt(rated, rtb, (size_t)II * DD);
  cvt(cand,  cdb, (size_t)BQ * DD);
  cvt(ie_w1, w1i, 512 * 512);   cvt(ie_w2, w2i, 512 * 256);
  cvt(ue_w1, w1u, 512 * 1024);  cvt(ue_w2, w2u, 1024 * 512);
  cvt(m_w1,  mw1, 768 * 1024);  cvt(m_w2,  mw2, 1024 * 512);
  cvt(m_w3,  mw3, 512 * 256);   cvt(m_w4,  mw4, 256 * 128);

  // 3) masked softmax * user_matrix -> Wb (bf16)
  attn_weights_kernel<<<BQ, 256, 0, stream>>>(cs, rs, um, b_att, Wb);

  // 4) user_est = Wb @ rated         (1024 x 512 x 4096)
  gemm_bf16_wmma<<<dim3(512 / BN, BQ / BM), 256, 0, stream>>>(
      Wb, rtb, ue, nullptr, 512, II, 512, 0);

  // 5) item tower
  gemm_bf16_wmma<<<dim3(512 / BN, BQ / BM), 256, 0, stream>>>(
      cdb, w1i, it1, ie_b1, 512, 512, 512, 1);
  gemm_bf16_wmma<<<dim3(256 / BN, BQ / BM), 256, 0, stream>>>(
      it1, w2i, hcat, ie_b2, 256, 512, 768, 1);               // -> hcat[:, 0:256]

  // 6) user tower
  gemm_bf16_wmma<<<dim3(1024 / BN, BQ / BM), 256, 0, stream>>>(
      ue, w1u, ut1, ue_b1, 1024, 512, 1024, 1);
  gemm_bf16_wmma<<<dim3(512 / BN, BQ / BM), 256, 0, stream>>>(
      ut1, w2u, hcat + 256, ue_b2, 512, 1024, 768, 1);        // -> hcat[:, 256:768]

  // 7) MLP trunk
  gemm_bf16_wmma<<<dim3(1024 / BN, BQ / BM), 256, 0, stream>>>(
      hcat, mw1, h1, m_b1, 1024, 768, 1024, 1);
  gemm_bf16_wmma<<<dim3(512 / BN, BQ / BM), 256, 0, stream>>>(
      h1, mw2, h2, m_b2, 512, 1024, 512, 1);
  gemm_bf16_wmma<<<dim3(256 / BN, BQ / BM), 256, 0, stream>>>(
      h2, mw3, h3, m_b3, 256, 512, 256, 1);
  gemm_bf16_wmma<<<dim3(128 / BN, BQ / BM), 256, 0, stream>>>(
      h3, mw4, h4, m_b4, 128, 256, 128, 1);

  // 8) final D4 -> 1
  rowdot_bf16_kernel<<<BQ, 32, 0, stream>>>(h4, m_w5, m_b5, out, 128);
}
